// EpisodicNovelty_25589415149739
// MI455X (gfx1250) — compile-verified, matrix operand order checked by reading, not run on hardware
//
#include <hip/hip_runtime.h>
#include <hip/hip_bf16.h>
#include <math.h>

typedef __attribute__((ext_vector_type(16))) __bf16 v16bf;
typedef __attribute__((ext_vector_type(8)))  __bf16 v8bf;
typedef __attribute__((ext_vector_type(8)))  float  v8f;

#define Q        256
#define D        512
#define NMEM     200000
#define K5       5
#define ROWS_PB  512                       // memory rows per block
#define GROUP    16                        // rows per WMMA B-tile group
#define NGROUPS  (ROWS_PB / GROUP)         // 32
#define NBLOCKS  ((NMEM + ROWS_PB - 1) / ROWS_PB)  // 391
#define BSTR     520                       // bf16 row stride (padded, bank-friendly)
#define SSTR     17                        // stage row stride in floats (padded)

__device__ __forceinline__ void insert5(float v, int gi,
                                        float& d0, float& d1, float& d2, float& d3, float& d4,
                                        int& i0, int& i1, int& i2, int& i3, int& i4) {
  if (v < d4) {
    d4 = v; i4 = gi;
    if (d4 < d3) { float t = d4; d4 = d3; d3 = t; int u = i4; i4 = i3; i3 = u; }
    if (d3 < d2) { float t = d3; d3 = d2; d2 = t; int u = i3; i3 = i2; i2 = u; }
    if (d2 < d1) { float t = d2; d2 = d1; d1 = t; int u = i2; i2 = i1; i1 = u; }
    if (d1 < d0) { float t = d1; d1 = d0; d0 = t; int u = i1; i1 = i0; i0 = u; }
  }
}

// Build a v16bf WMMA operand from 16 fp32 values at p[0..7] and p[16..23].
__device__ __forceinline__ v16bf make_op(const float* p) {
  float4 x0 = *(const float4*)(p);
  float4 x1 = *(const float4*)(p + 4);
  float4 y0 = *(const float4*)(p + 16);
  float4 y1 = *(const float4*)(p + 20);
  v16bf a;
  a[0]  = (__bf16)x0.x; a[1]  = (__bf16)x0.y; a[2]  = (__bf16)x0.z; a[3]  = (__bf16)x0.w;
  a[4]  = (__bf16)x1.x; a[5]  = (__bf16)x1.y; a[6]  = (__bf16)x1.z; a[7]  = (__bf16)x1.w;
  a[8]  = (__bf16)y0.x; a[9]  = (__bf16)y0.y; a[10] = (__bf16)y0.z; a[11] = (__bf16)y0.w;
  a[12] = (__bf16)y1.x; a[13] = (__bf16)y1.y; a[14] = (__bf16)y1.z; a[15] = (__bf16)y1.w;
  return a;
}

// ---------------------------------------------------------------------------
// Phase 1: bf16 WMMA distance GEMM + per-block top-5 per query.
// Block = 256 threads = 8 wave32. A (all 256 queries) lives in REGISTERS
// (loop-invariant across row groups): 0.5 KB LDS traffic per WMMA, B shared
// by both M-tiles. B global loads software-pipelined one group ahead.
// ---------------------------------------------------------------------------
__global__ __launch_bounds__(256, 1)
void knn_phase1(const float* __restrict__ emb, const float* __restrict__ mem,
                float* __restrict__ cand_d, int* __restrict__ cand_i) {
  __shared__ __bf16 Bbuf[GROUP * BSTR];      // 16,640 B
  __shared__ float  sqq[Q];                  //  1,024 B
  __shared__ float  sqm[GROUP];              //     64 B
  __shared__ float  stage[16 * 16 * SSTR];   // 17,408 B

  const int tid  = threadIdx.x;
  const int wave = tid >> 5;
  const int lane = tid & 31;
  const int kb   = (lane < 16) ? 0 : 8;     // per-lane K sub-offset (WMMA layout)

  // ---- ||q||^2 (one query row per thread, fp32, L2-hot)
  {
    const float* row = emb + (size_t)tid * D;
    float s = 0.f;
#pragma unroll 8
    for (int c = 0; c < D; c += 4) {
      float4 v = *(const float4*)(row + c);
      s += v.x * v.x + v.y * v.y + v.z * v.z + v.w * v.w;
    }
    sqq[tid] = s;
  }

  // ---- Preload A operands into registers (invariant across all row groups).
  // Tile t2 covers queries (wave*2+t2)*16 .. +15 ; lane holds row q0 + lane%16.
  v16bf areg0[16], areg1[16];
  {
    const float* r0 = emb + (size_t)(wave * 32 +      (lane & 15)) * D + kb;
    const float* r1 = emb + (size_t)(wave * 32 + 16 + (lane & 15)) * D + kb;
#pragma unroll
    for (int k0 = 0; k0 < 16; ++k0) {
      areg0[k0] = make_op(r0 + k0 * 32);
      areg1[k0] = make_op(r1 + k0 * 32);
    }
  }

  // per-lane top-5 (lane owns query wave*32 + lane)
  float d0 = 1e38f, d1 = 1e38f, d2 = 1e38f, d3 = 1e38f, d4 = 1e38f;
  int   i0 = 0, i1 = 0, i2 = 0, i3 = 0, i4 = 0;

  const int rowBlock = blockIdx.x * ROWS_PB;
  const int rB  = tid >> 4;          // B-load: row within group
  const int cB  = (tid & 15) * 32;   // B-load: 32 contiguous cols per thread

  // ---- Pipeline prologue: fetch group 0 fp32 into registers.
  float4 f[8];
  {
    const int grow = rowBlock + rB;
    if (grow < NMEM) {
      const float* src = mem + (size_t)grow * D + cB;
#pragma unroll
      for (int i = 0; i < 8; ++i) f[i] = *(const float4*)(src + i * 4);
    } else {
#pragma unroll
      for (int i = 0; i < 8; ++i) f[i] = make_float4(0.f, 0.f, 0.f, 0.f);
    }
  }

  for (int g = 0; g < NGROUPS; ++g) {
    const int rowsBase = rowBlock + g * GROUP;
    __syncthreads();  // previous group fully consumed (Bbuf/sqm/stage reusable)

    // ---- Convert prefetched fp32 -> bf16 LDS, fused ||m||^2.
    {
      float s = 0.f;
      __bf16* dst = &Bbuf[rB * BSTR + cB];
#pragma unroll
      for (int i = 0; i < 8; ++i) {
        float4 v = f[i];
        s += v.x * v.x + v.y * v.y + v.z * v.z + v.w * v.w;
        dst[i * 4 + 0] = (__bf16)v.x; dst[i * 4 + 1] = (__bf16)v.y;
        dst[i * 4 + 2] = (__bf16)v.z; dst[i * 4 + 3] = (__bf16)v.w;
      }
#pragma unroll
      for (int off = 8; off > 0; off >>= 1) s += __shfl_xor(s, off, 16);
      if ((lane & 15) == 0) sqm[rB] = ((rowsBase + rB) < NMEM) ? s : 1e30f;
    }

    // ---- Issue next group's global loads now; they complete during compute.
    if (g + 1 < NGROUPS) {
      const int grow = rowsBase + GROUP + rB;
      if (grow < NMEM) {
        const float* src = mem + (size_t)grow * D + cB;
#pragma unroll
        for (int i = 0; i < 8; ++i) f[i] = *(const float4*)(src + i * 4);
      } else {
#pragma unroll
        for (int i = 0; i < 8; ++i) f[i] = make_float4(0.f, 0.f, 0.f, 0.f);
      }
    }
    __syncthreads();

    // ---- WMMA: B loaded once per K-step, shared by both M-tiles.
    const int nrow = lane & 15;
    v8f c0 = {}, c1 = {};
    {
      const __bf16* bB = &Bbuf[nrow * BSTR + kb];
#pragma unroll
      for (int k0 = 0; k0 < 16; ++k0) {
        v8bf blo = *(const v8bf*)(bB + k0 * 32);
        v8bf bhi = *(const v8bf*)(bB + k0 * 32 + 16);
        v16bf b = __builtin_shufflevector(blo, bhi,
                    0,1,2,3,4,5,6,7,8,9,10,11,12,13,14,15);
        c0 = __builtin_amdgcn_wmma_f32_16x16x32_bf16(false, areg0[k0], false, b,
                                                     (short)0, c0, false, false);
        c1 = __builtin_amdgcn_wmma_f32_16x16x32_bf16(false, areg1[k0], false, b,
                                                     (short)0, c1, false, false);
      }
    }

    // ---- d2 = ||q||^2 + ||m||^2 - 2 q.m -> stage tiles.
    {
      const int mbase = (lane < 16) ? 0 : 8;
      const float sm = sqm[nrow];
      float* st0 = &stage[(wave * 2 + 0) * 16 * SSTR];
      float* st1 = &stage[(wave * 2 + 1) * 16 * SSTR];
      const int q0 = wave * 32, q1 = wave * 32 + 16;
#pragma unroll
      for (int r = 0; r < 8; ++r) {
        const int m = mbase + r;
        st0[m * SSTR + nrow] = sqq[q0 + m] + sm - 2.0f * c0[r];
        st1[m * SSTR + nrow] = sqq[q1 + m] + sm - 2.0f * c1[r];
      }
    }

    // ---- Selection: lane owns one query; scan its 16 candidates.
    // Same-wave LDS RAW: DS ops are in-order per wave (DScnt waits inserted).
    {
      const float* st = &stage[((wave * 2 + (lane >> 4)) * 16 + (lane & 15)) * SSTR];
#pragma unroll
      for (int n = 0; n < 16; ++n)
        insert5(st[n], rowsBase + n, d0, d1, d2, d3, d4, i0, i1, i2, i3, i4);
    }
  }

  // ---- Emit per-block top-5 for this lane's query.
  const int qq = wave * 32 + lane;
  const size_t base = ((size_t)blockIdx.x * Q + qq) * K5;
  cand_d[base + 0] = d0; cand_d[base + 1] = d1; cand_d[base + 2] = d2;
  cand_d[base + 3] = d3; cand_d[base + 4] = d4;
  cand_i[base + 0] = i0; cand_i[base + 1] = i1; cand_i[base + 2] = i2;
  cand_i[base + 3] = i3; cand_i[base + 4] = i4;
}

// ---------------------------------------------------------------------------
// Phase 2: per-query merge of NBLOCKS*5 candidates -> global top-5, then
// exact fp32 distances (matches reference recompute) -> per-query sum.
// ---------------------------------------------------------------------------
__global__ __launch_bounds__(64)
void knn_phase2(const float* __restrict__ emb, const float* __restrict__ mem,
                const float* __restrict__ cand_d, const int* __restrict__ cand_i,
                float* __restrict__ qsum) {
  const int q = blockIdx.x;
  const int t = threadIdx.x;
  const int NC = NBLOCKS * K5;

  float d0 = 1e38f, d1 = 1e38f, d2 = 1e38f, d3 = 1e38f, d4 = 1e38f;
  int   i0 = 0, i1 = 0, i2 = 0, i3 = 0, i4 = 0;
  for (int c = t; c < NC; c += 64) {
    const int b = c / K5, j = c - b * K5;
    const size_t off = ((size_t)b * Q + q) * K5 + j;
    insert5(cand_d[off], cand_i[off], d0, d1, d2, d3, d4, i0, i1, i2, i3, i4);
  }

  __shared__ float sd[64 * K5];
  __shared__ int   si[64 * K5];
  sd[t * K5 + 0] = d0; sd[t * K5 + 1] = d1; sd[t * K5 + 2] = d2;
  sd[t * K5 + 3] = d3; sd[t * K5 + 4] = d4;
  si[t * K5 + 0] = i0; si[t * K5 + 1] = i1; si[t * K5 + 2] = i2;
  si[t * K5 + 3] = i3; si[t * K5 + 4] = i4;
  __syncthreads();

  __shared__ int sel[K5];
  if (t == 0) {
    float e0 = 1e38f, e1 = 1e38f, e2 = 1e38f, e3 = 1e38f, e4 = 1e38f;
    int   j0 = 0, j1 = 0, j2 = 0, j3 = 0, j4 = 0;
    for (int c = 0; c < 64 * K5; ++c)
      insert5(sd[c], si[c], e0, e1, e2, e3, e4, j0, j1, j2, j3, j4);
    sel[0] = j0; sel[1] = j1; sel[2] = j2; sel[3] = j3; sel[4] = j4;
  }
  __syncthreads();

  __shared__ float red[64];
  float acc = 0.f;
  const float* qrow = emb + (size_t)q * D;
  for (int j = 0; j < K5; ++j) {
    const float* mrow = mem + (size_t)sel[j] * D;
    float s = 0.f;
    for (int c = t; c < D; c += 64) { const float df = mrow[c] - qrow[c]; s += df * df; }
    red[t] = s;
    __syncthreads();
    if (t == 0) {
      float tot = 0.f;
      for (int x = 0; x < 64; ++x) tot += red[x];
      acc += sqrtf(tot);
    }
    __syncthreads();
  }
  if (t == 0) qsum[q] = acc;
}

// ---------------------------------------------------------------------------
// Phase 3: deterministic mean over 256 per-query sums.
// ---------------------------------------------------------------------------
__global__ __launch_bounds__(256)
void knn_phase3(const float* __restrict__ qsum, float* __restrict__ out) {
  __shared__ float red[Q];
  const int t = threadIdx.x;
  red[t] = qsum[t];
  __syncthreads();
  for (int o = Q / 2; o > 0; o >>= 1) {
    if (t < o) red[t] += red[t + o];
    __syncthreads();
  }
  if (t == 0) out[0] = red[0] / (float)(Q * K5);
}

extern "C" void kernel_launch(void* const* d_in, const int* in_sizes, int n_in,
                              void* d_out, int out_size, void* d_ws, size_t ws_size,
                              hipStream_t stream) {
  const float* emb = (const float*)d_in[0];   // (256, 512) fp32
  const float* mem = (const float*)d_in[1];   // (200000, 512) fp32
  float* out = (float*)d_out;

  char* ws = (char*)d_ws;
  const size_t cdBytes = (size_t)NBLOCKS * Q * K5 * sizeof(float);  // ~2.0 MB
  float* cand_d = (float*)ws;
  int*   cand_i = (int*)(ws + cdBytes);
  float* qsum   = (float*)(ws + 2 * cdBytes);

  knn_phase1<<<NBLOCKS, 256, 0, stream>>>(emb, mem, cand_d, cand_i);
  knn_phase2<<<Q, 64, 0, stream>>>(emb, mem, cand_d, cand_i, qsum);
  knn_phase3<<<1, Q, 0, stream>>>(qsum, out);
}
